// GAT_G_70961449664986
// MI455X (gfx1250) — matrix operand order repaired and checked by reading, not compiled, not run on hardware
//
#include <hip/hip_runtime.h>
#include <hip/hip_bf16.h>
#include <math.h>

typedef __attribute__((ext_vector_type(16))) _Float16 v16h;
typedef __attribute__((ext_vector_type(8)))  _Float16 v8h;
typedef __attribute__((ext_vector_type(8)))  float    v8f;

#define NEG_SLOPE 0.2f
#define ORD_NEG_INF 0x007fffffu   // f2ord(-inf)

static __device__ __forceinline__ unsigned f2ord(float f) {
  unsigned u = __float_as_uint(f);
  return (u & 0x80000000u) ? ~u : (u | 0x80000000u);
}
static __device__ __forceinline__ float ord2f(unsigned u) {
  u = (u & 0x80000000u) ? (u & 0x7fffffffu) : ~u;
  return __uint_as_float(u);
}
static __device__ __forceinline__ float lrelu(float v) {
  return v > 0.0f ? v : v * NEG_SLOPE;
}

// ---------------------------------------------------------------- fills
__global__ void k_fill_u32(unsigned* __restrict__ p, unsigned v, long long n) {
  long long i = (long long)blockIdx.x * blockDim.x + threadIdx.x;
  long long stride = (long long)gridDim.x * blockDim.x;
  for (; i < n; i += stride) p[i] = v;
}

// ---------------------------------------------------------------- weight prep
// W1t: [80][32] f16, (cc, K). cc 0..63 = W1 columns, cc 64..71 = W1·att_src1,
//      cc 72..79 = W1·att_dst1. K 16..31 zero-padded.
// W2t: [16][64] f16, (col, K). col 0..7 = W2 columns, col 8 = W2·att_src2,
//      col 9 = W2·att_dst2, col 10..15 = 0.
__global__ void k_cvt_weights(const float* __restrict__ W1, const float* __restrict__ as1,
                              const float* __restrict__ ad1,
                              const float* __restrict__ W2, const float* __restrict__ as2,
                              const float* __restrict__ ad2,
                              _Float16* __restrict__ W1t, _Float16* __restrict__ W2t) {
  for (int idx = threadIdx.x; idx < 80 * 32; idx += blockDim.x) {
    int cc = idx >> 5, K = idx & 31;
    float v = 0.0f;
    if (K < 16) {
      if (cc < 64) {
        v = W1[K * 64 + cc];
      } else if (cc < 72) {
        int h = cc - 64;
        for (int c = 0; c < 8; ++c) v += W1[K * 64 + h * 8 + c] * as1[h * 8 + c];
      } else {
        int h = cc - 72;
        for (int c = 0; c < 8; ++c) v += W1[K * 64 + h * 8 + c] * ad1[h * 8 + c];
      }
    }
    W1t[idx] = (_Float16)v;
  }
  for (int idx = threadIdx.x; idx < 16 * 64; idx += blockDim.x) {
    int col = idx >> 6, k = idx & 63;
    float v = 0.0f;
    if (col < 8) {
      v = W2[k * 8 + col];
    } else if (col == 8) {
      for (int c = 0; c < 8; ++c) v += W2[k * 8 + c] * as2[c];
    } else if (col == 9) {
      for (int c = 0; c < 8; ++c) v += W2[k * 8 + c] * ad2[c];
    }
    W2t[idx] = (_Float16)v;
  }
}

// ---------------------------------------------------------------- conv1 projection:
// [h1 | es1 | ed1] = x[N,16] @ [W1 | W1·as1 | W1·ad1]   (WMMA f16, K padded 16->32)
__global__ void k_proj1(const float* __restrict__ x, const _Float16* __restrict__ W1t,
                        float* __restrict__ h1, float* __restrict__ es1,
                        float* __restrict__ ed1, int n_tiles, int N) {
  __shared__ _Float16 sW[80 * 32];
  for (int i = threadIdx.x; i < 80 * 32 / 2; i += blockDim.x)
    ((unsigned*)sW)[i] = ((const unsigned*)W1t)[i];
  __syncthreads();

  int wave = (int)((blockIdx.x * blockDim.x + threadIdx.x) >> 5);
  int lane = threadIdx.x & 31;
  if (wave >= n_tiles) return;            // wave-uniform; EXEC all-ones at WMMA
  int hi  = lane >> 4;
  int col = lane & 15;
  int row = wave * 16 + (lane & 15);
  int rowc = row < N ? row : N - 1;
  int row0 = wave * 16 + hi * 8;          // first output row this lane touches

  // A: lane element i -> K = (i>=8?16:0) + hi*8 + (i&7); real K<16 only.
  float4 x0 = *(const float4*)(x + (size_t)rowc * 16 + hi * 8);
  float4 x1 = *(const float4*)(x + (size_t)rowc * 16 + hi * 8 + 4);
  float xv[8] = {x0.x, x0.y, x0.z, x0.w, x1.x, x1.y, x1.z, x1.w};
  v16h a;
#pragma unroll
  for (int i = 0; i < 8; ++i) { a[i] = (_Float16)xv[i]; a[i + 8] = (_Float16)0.0f; }

  const bool full = (wave * 16 + 16) <= N;
#pragma unroll
  for (int c = 0; c < 5; ++c) {
    int cc = c * 16 + col;
    // B: lane element i -> K = (i>=8?16:0) + hi*8 + (i&7)  (zero-padded K in LDS => no selects)
    const _Float16* bp = sW + cc * 32 + hi * 8;
    v8h b0 = *(const v8h*)bp;
    v8h b1 = *(const v8h*)(bp + 16);
    v16h b;
#pragma unroll
    for (int i = 0; i < 8; ++i) { b[i] = b0[i]; b[i + 8] = b1[i]; }

    v8f acc = {};
    acc = __builtin_amdgcn_wmma_f32_16x16x32_f16(false, a, false, b, (short)0, acc, false, false);

    if (c < 4) {
      float* out0 = h1 + (size_t)row0 * 64 + c * 16 + col;   // row stride 64 floats
      if (full) {
#pragma unroll
        for (int r = 0; r < 8; ++r) out0[r * 64] = acc[r];
      } else {
#pragma unroll
        for (int r = 0; r < 8; ++r)
          if (row0 + r < N) out0[r * 64] = acc[r];
      }
    } else {  // attention-coefficient columns: 0..7 -> es1, 8..15 -> ed1
      float* dst = (col < 8) ? es1 : ed1;
      float* out0 = dst + (size_t)row0 * 8 + (col & 7);      // row stride 8 floats
      if (full) {
#pragma unroll
        for (int r = 0; r < 8; ++r) out0[r * 8] = acc[r];
      } else {
#pragma unroll
        for (int r = 0; r < 8; ++r)
          if (row0 + r < N) out0[r * 8] = acc[r];
      }
    }
  }
}

// ---------------------------------------------------------------- conv1 edge passes
__global__ void k_edge1_max(const int* __restrict__ ei, int E, int N,
                            const float* __restrict__ es, const float* __restrict__ ed,
                            unsigned* __restrict__ m) {
  int e = blockIdx.x * blockDim.x + threadIdx.x;
  if (e >= E + N) return;
  int s = (e < E) ? ei[e] : (e - E);
  int d = (e < E) ? ei[E + e] : (e - E);
  float4 s0 = *(const float4*)(es + (size_t)s * 8), s1 = *(const float4*)(es + (size_t)s * 8 + 4);
  float4 d0 = *(const float4*)(ed + (size_t)d * 8), d1 = *(const float4*)(ed + (size_t)d * 8 + 4);
  float sv[8] = {s0.x, s0.y, s0.z, s0.w, s1.x, s1.y, s1.z, s1.w};
  float dv[8] = {d0.x, d0.y, d0.z, d0.w, d1.x, d1.y, d1.z, d1.w};
#pragma unroll
  for (int h = 0; h < 8; ++h)
    atomicMax(&m[d * 8 + h], f2ord(lrelu(sv[h] + dv[h])));
}
__global__ void k_edge1_sum(const int* __restrict__ ei, int E, int N,
                            const float* __restrict__ es, const float* __restrict__ ed,
                            const unsigned* __restrict__ m, float* __restrict__ dn) {
  int e = blockIdx.x * blockDim.x + threadIdx.x;
  if (e >= E + N) return;
  int s = (e < E) ? ei[e] : (e - E);
  int d = (e < E) ? ei[E + e] : (e - E);
  float4 s0 = *(const float4*)(es + (size_t)s * 8), s1 = *(const float4*)(es + (size_t)s * 8 + 4);
  float4 d0 = *(const float4*)(ed + (size_t)d * 8), d1 = *(const float4*)(ed + (size_t)d * 8 + 4);
  float sv[8] = {s0.x, s0.y, s0.z, s0.w, s1.x, s1.y, s1.z, s1.w};
  float dv[8] = {d0.x, d0.y, d0.z, d0.w, d1.x, d1.y, d1.z, d1.w};
#pragma unroll
  for (int h = 0; h < 8; ++h) {
    float v = lrelu(sv[h] + dv[h]);
    atomicAdd(&dn[d * 8 + h], __expf(v - ord2f(m[d * 8 + h])));
  }
}
__global__ void k_edge1_agg(const int* __restrict__ ei, int E, int N,
                            const float* __restrict__ es, const float* __restrict__ ed,
                            const unsigned* __restrict__ m, const float* __restrict__ dn,
                            const float* __restrict__ h1, float* __restrict__ agg) {
  int e = blockIdx.x * blockDim.x + threadIdx.x;
  if (e >= E + N) return;
  int s = (e < E) ? ei[e] : (e - E);
  int d = (e < E) ? ei[E + e] : (e - E);
  float4 s0 = *(const float4*)(es + (size_t)s * 8), s1 = *(const float4*)(es + (size_t)s * 8 + 4);
  float4 d0 = *(const float4*)(ed + (size_t)d * 8), d1 = *(const float4*)(ed + (size_t)d * 8 + 4);
  float sv[8] = {s0.x, s0.y, s0.z, s0.w, s1.x, s1.y, s1.z, s1.w};
  float dv[8] = {d0.x, d0.y, d0.z, d0.w, d1.x, d1.y, d1.z, d1.w};
#pragma unroll
  for (int h = 0; h < 8; ++h) {
    float v = lrelu(sv[h] + dv[h]);
    float alpha = __expf(v - ord2f(m[d * 8 + h])) / dn[d * 8 + h];
    float4 f0 = *(const float4*)(h1 + (size_t)s * 64 + h * 8);
    float4 f1 = *(const float4*)(h1 + (size_t)s * 64 + h * 8 + 4);
    float fv[8] = {f0.x, f0.y, f0.z, f0.w, f1.x, f1.y, f1.z, f1.w};
#pragma unroll
    for (int c = 0; c < 8; ++c)
      atomicAdd(&agg[(size_t)d * 64 + h * 8 + c], fv[c] * alpha);
  }
}

// ---------------------------------------------------------------- bias + ELU -> f16 features for conv2
__global__ void k_elu(const float* __restrict__ agg1, const float* __restrict__ b1,
                      _Float16* __restrict__ x2h, long long n) {
  long long t = (long long)blockIdx.x * blockDim.x + threadIdx.x;
  if (t >= n) return;
  float v = agg1[t] + b1[t & 63];
  v = v > 0.f ? v : (__expf(v) - 1.f);
  x2h[t] = (_Float16)v;
}

// ---------------------------------------------------------------- conv2 projection:
// [h2 | es2 | ed2] = x2[N,64] @ [W2 | W2·as2 | W2·ad2]  (WMMA, 2 K-steps)
__global__ void k_proj2(const _Float16* __restrict__ x2h, const _Float16* __restrict__ W2t,
                        float* __restrict__ h2, float* __restrict__ es2,
                        float* __restrict__ ed2, int n_tiles, int N) {
  __shared__ _Float16 sW[16 * 64];
  for (int i = threadIdx.x; i < 16 * 64 / 2; i += blockDim.x)
    ((unsigned*)sW)[i] = ((const unsigned*)W2t)[i];
  __syncthreads();

  int wave = (int)((blockIdx.x * blockDim.x + threadIdx.x) >> 5);
  int lane = threadIdx.x & 31;
  if (wave >= n_tiles) return;
  int hi  = lane >> 4;
  int col = lane & 15;
  int row = wave * 16 + (lane & 15);
  int rowc = row < N ? row : N - 1;
  int row0 = wave * 16 + hi * 8;
  const _Float16* xr = x2h + (size_t)rowc * 64;

  v8f acc = {};
#pragma unroll
  for (int s = 0; s < 2; ++s) {
    // A: element i -> K = 32s + (i>=8?16:0) + hi*8 + (i&7)  => two contiguous 16B loads
    const _Float16* ap = xr + 32 * s + hi * 8;
    v8h a0 = *(const v8h*)ap;
    v8h a1 = *(const v8h*)(ap + 16);
    v16h a;
#pragma unroll
    for (int i = 0; i < 8; ++i) { a[i] = a0[i]; a[i + 8] = a1[i]; }
    // B: element i -> K = 32s + (i>=8?16:0) + hi*8 + (i&7)  => two 16B loads from LDS
    const _Float16* bp = sW + col * 64 + 32 * s + hi * 8;
    v8h b0 = *(const v8h*)bp;
    v8h b1 = *(const v8h*)(bp + 16);
    v16h b;
#pragma unroll
    for (int i = 0; i < 8; ++i) { b[i] = b0[i]; b[i + 8] = b1[i]; }
    acc = __builtin_amdgcn_wmma_f32_16x16x32_f16(false, a, false, b, (short)0, acc, false, false);
  }

  const bool full = (wave * 16 + 16) <= N;
  if (col < 8) {
    float* out0 = h2 + (size_t)row0 * 8 + col;     // row stride 8 floats
    if (full) {
#pragma unroll
      for (int r = 0; r < 8; ++r) out0[r * 8] = acc[r];
    } else {
#pragma unroll
      for (int r = 0; r < 8; ++r)
        if (row0 + r < N) out0[r * 8] = acc[r];
    }
  } else if (col < 10) {
    float* dst = (col == 8) ? es2 : ed2;
    float* out0 = dst + row0;                      // row stride 1 float
    if (full) {
#pragma unroll
      for (int r = 0; r < 8; ++r) out0[r] = acc[r];
    } else {
#pragma unroll
      for (int r = 0; r < 8; ++r)
        if (row0 + r < N) out0[r] = acc[r];
    }
  }
}

// ---------------------------------------------------------------- conv2 edge passes (1 head)
__global__ void k_edge2_max(const int* __restrict__ ei, int E, int N,
                            const float* __restrict__ es, const float* __restrict__ ed,
                            unsigned* __restrict__ m) {
  int e = blockIdx.x * blockDim.x + threadIdx.x;
  if (e >= E + N) return;
  int s = (e < E) ? ei[e] : (e - E);
  int d = (e < E) ? ei[E + e] : (e - E);
  atomicMax(&m[d], f2ord(lrelu(es[s] + ed[d])));
}
__global__ void k_edge2_sum(const int* __restrict__ ei, int E, int N,
                            const float* __restrict__ es, const float* __restrict__ ed,
                            const unsigned* __restrict__ m, float* __restrict__ dn) {
  int e = blockIdx.x * blockDim.x + threadIdx.x;
  if (e >= E + N) return;
  int s = (e < E) ? ei[e] : (e - E);
  int d = (e < E) ? ei[E + e] : (e - E);
  float v = lrelu(es[s] + ed[d]);
  atomicAdd(&dn[d], __expf(v - ord2f(m[d])));
}
__global__ void k_edge2_agg(const int* __restrict__ ei, int E, int N,
                            const float* __restrict__ es, const float* __restrict__ ed,
                            const unsigned* __restrict__ m, const float* __restrict__ dn,
                            const float* __restrict__ h2, float* __restrict__ agg2) {
  int e = blockIdx.x * blockDim.x + threadIdx.x;
  if (e >= E + N) return;
  int s = (e < E) ? ei[e] : (e - E);
  int d = (e < E) ? ei[E + e] : (e - E);
  float v = lrelu(es[s] + ed[d]);
  float alpha = __expf(v - ord2f(m[d])) / dn[d];
  float4 f0 = *(const float4*)(h2 + (size_t)s * 8);
  float4 f1 = *(const float4*)(h2 + (size_t)s * 8 + 4);
  float fv[8] = {f0.x, f0.y, f0.z, f0.w, f1.x, f1.y, f1.z, f1.w};
#pragma unroll
  for (int c = 0; c < 8; ++c)
    atomicAdd(&agg2[(size_t)d * 8 + c], fv[c] * alpha);
}

// ---------------------------------------------------------------- global mean pool (sum + count)
__global__ void k_pool(const int* __restrict__ batch, const float* __restrict__ agg2,
                       const float* __restrict__ b2,
                       float* __restrict__ pooled, float* __restrict__ cnt, int N) {
  int n = blockIdx.x * blockDim.x + threadIdx.x;
  if (n >= N) return;
  int g = batch[n];
  float4 f0 = *(const float4*)(agg2 + (size_t)n * 8);
  float4 f1 = *(const float4*)(agg2 + (size_t)n * 8 + 4);
  float fv[8] = {f0.x, f0.y, f0.z, f0.w, f1.x, f1.y, f1.z, f1.w};
#pragma unroll
  for (int c = 0; c < 8; ++c)
    atomicAdd(&pooled[g * 8 + c], fv[c] + b2[c]);
  atomicAdd(&cnt[g], 1.0f);
}

// ---------------------------------------------------------------- MLP head + log_softmax
__global__ void k_head(const float* __restrict__ pooled, const float* __restrict__ cnt,
                       const float* __restrict__ stats,
                       const float* __restrict__ fc1W, const float* __restrict__ fc1b,
                       const float* __restrict__ fc2W, const float* __restrict__ fc2b,
                       float* __restrict__ out, int G) {
  int g = blockIdx.x * blockDim.x + threadIdx.x;
  if (g >= G) return;
  float feat[18];
  float c = fmaxf(cnt[g], 1.0f);
#pragma unroll
  for (int i = 0; i < 8; ++i) feat[i] = pooled[g * 8 + i] / c;
#pragma unroll
  for (int i = 0; i < 10; ++i) feat[8 + i] = stats[g * 10 + i];
  float z1[20];
#pragma unroll
  for (int j = 0; j < 20; ++j) {
    float a = fc1b[j];
#pragma unroll
    for (int i = 0; i < 18; ++i) a += feat[i] * fc1W[i * 20 + j];
    z1[j] = a > 0.f ? a : 0.f;
  }
  float z2[5], mx = -1e30f;
#pragma unroll
  for (int j = 0; j < 5; ++j) {
    float a = fc2b[j];
#pragma unroll
    for (int i = 0; i < 20; ++i) a += z1[i] * fc2W[i * 5 + j];
    z2[j] = a; mx = fmaxf(mx, a);
  }
  float se = 0.f;
#pragma unroll
  for (int j = 0; j < 5; ++j) se += __expf(z2[j] - mx);
  float lse = mx + __logf(se);
#pragma unroll
  for (int j = 0; j < 5; ++j) out[g * 5 + j] = z2[j] - lse;
}

// ================================================================ launcher
extern "C" void kernel_launch(void* const* d_in, const int* in_sizes, int n_in,
                              void* d_out, int out_size, void* d_ws, size_t ws_size,
                              hipStream_t stream) {
  const float* x     = (const float*)d_in[0];
  const int*   ei    = (const int*)d_in[1];
  const int*   batch = (const int*)d_in[2];
  // d_in[3] (eig) is unused by the reference
  const float* stats = (const float*)d_in[4];
  const float* W1    = (const float*)d_in[5];
  const float* as1   = (const float*)d_in[6];
  const float* ad1   = (const float*)d_in[7];
  const float* b1    = (const float*)d_in[8];
  const float* W2    = (const float*)d_in[9];
  const float* as2   = (const float*)d_in[10];
  const float* ad2   = (const float*)d_in[11];
  const float* b2    = (const float*)d_in[12];
  const float* fc1W  = (const float*)d_in[13];
  const float* fc1b  = (const float*)d_in[14];
  const float* fc2W  = (const float*)d_in[15];
  const float* fc2b  = (const float*)d_in[16];
  float* out = (float*)d_out;

  const int N = in_sizes[0] / 16;
  const int E = in_sizes[1] / 2;
  const int G = in_sizes[4] / 10;
  const int Etot = E + N;
  const int ntiles = (N + 15) / 16;

  // ---- workspace carve-up (each chunk 256B aligned) ----
  char* base = (char*)d_ws;
  size_t off = 0;
  auto alloc = [&](size_t bytes) -> void* {
    void* r = base + off;
    off = (off + bytes + 255) & ~(size_t)255;
    return r;
  };
  _Float16* W1t = (_Float16*)alloc(80 * 32 * sizeof(_Float16));
  _Float16* W2t = (_Float16*)alloc(16 * 64 * sizeof(_Float16));
  float* h1  = (float*)alloc((size_t)N * 64 * sizeof(float));
  float* es1 = (float*)alloc((size_t)N * 8 * sizeof(float));
  float* ed1 = (float*)alloc((size_t)N * 8 * sizeof(float));
  float* h2  = (float*)alloc((size_t)N * 8 * sizeof(float));
  float* es2 = (float*)alloc((size_t)N * sizeof(float));
  float* ed2 = (float*)alloc((size_t)N * sizeof(float));
  _Float16* x2h = (_Float16*)alloc((size_t)N * 64 * sizeof(_Float16));
  // zero-initialized region (contiguous)
  size_t zstart = off;
  float* dn1    = (float*)alloc((size_t)N * 8 * sizeof(float));
  float* agg1   = (float*)alloc((size_t)N * 64 * sizeof(float));
  float* dn2    = (float*)alloc((size_t)N * sizeof(float));
  float* agg2   = (float*)alloc((size_t)N * 8 * sizeof(float));
  float* pooled = (float*)alloc((size_t)G * 8 * sizeof(float));
  float* cnt    = (float*)alloc((size_t)G * sizeof(float));
  size_t zend = off;
  // -inf (ordered-encoding) region (contiguous)
  size_t nstart = off;
  unsigned* m1 = (unsigned*)alloc((size_t)N * 8 * sizeof(unsigned));
  unsigned* m2 = (unsigned*)alloc((size_t)N * sizeof(unsigned));
  size_t nend = off;

  const int TB = 256;
  const int gE = (Etot + TB - 1) / TB;

  // init scratch (re-runs on every graph replay; deterministic)
  k_fill_u32<<<1024, TB, 0, stream>>>((unsigned*)(base + zstart), 0u,
                                      (long long)((zend - zstart) / 4));
  k_fill_u32<<<256, TB, 0, stream>>>((unsigned*)(base + nstart), ORD_NEG_INF,
                                     (long long)((nend - nstart) / 4));
  k_cvt_weights<<<1, 256, 0, stream>>>(W1, as1, ad1, W2, as2, ad2, W1t, W2t);

  // ---- conv1 ----
  k_proj1<<<(ntiles + 7) / 8, TB, 0, stream>>>(x, W1t, h1, es1, ed1, ntiles, N);
  k_edge1_max<<<gE, TB, 0, stream>>>(ei, E, N, es1, ed1, m1);
  k_edge1_sum<<<gE, TB, 0, stream>>>(ei, E, N, es1, ed1, m1, dn1);
  k_edge1_agg<<<gE, TB, 0, stream>>>(ei, E, N, es1, ed1, m1, dn1, h1, agg1);
  k_elu<<<(int)(((long long)N * 64 + TB - 1) / TB), TB, 0, stream>>>(agg1, b1, x2h,
                                                                     (long long)N * 64);
  // ---- conv2 ----
  k_proj2<<<(ntiles + 7) / 8, TB, 0, stream>>>(x2h, W2t, h2, es2, ed2, ntiles, N);
  k_edge2_max<<<gE, TB, 0, stream>>>(ei, E, N, es2, ed2, m2);
  k_edge2_sum<<<gE, TB, 0, stream>>>(ei, E, N, es2, ed2, m2, dn2);
  k_edge2_agg<<<gE, TB, 0, stream>>>(ei, E, N, es2, ed2, m2, dn2, h2, agg2);

  // ---- pool + head ----
  k_pool<<<(N + TB - 1) / TB, TB, 0, stream>>>(batch, agg2, b2, pooled, cnt, N);
  k_head<<<(G + TB - 1) / TB, TB, 0, stream>>>(pooled, cnt, stats, fc1W, fc1b,
                                               fc2W, fc2b, out, G);
}
